// LiveNet_87677462381133
// MI455X (gfx1250) — compile-verified, matrix operand order checked by reading, not run on hardware
//
#include <hip/hip_runtime.h>
#include <hip/hip_bf16.h>
#include <math.h>

typedef __attribute__((ext_vector_type(16))) _Float16 v16h;
typedef __attribute__((ext_vector_type(8)))  _Float16 v8h;
typedef __attribute__((ext_vector_type(8)))  float    v8f;

#define BB 32768
#define NUM_IN 20
#define H1 1024
#define H2 512
#define ROWS 32            // batch rows per workgroup (2 WMMA row-blocks)
#define R2C 0.2025f        // (0.2+0.2+0.05)^2

// ---------------------------------------------------------------------------
// Kernel 1: convert weights to f16 (W1 padded 20->32 along K)
// ---------------------------------------------------------------------------
__global__ void cvt_weights(const float* __restrict__ W1,
                            const float* __restrict__ W21,
                            const float* __restrict__ Wobs1,
                            _Float16* __restrict__ W1h,
                            _Float16* __restrict__ W21h,
                            _Float16* __restrict__ Wobs1h) {
    int t = blockIdx.x * 256 + threadIdx.x;
    if (t < H1 * 32) {
        int r = t >> 5, k = t & 31;
        W1h[t] = (k < NUM_IN) ? (_Float16)W1[r * NUM_IN + k] : (_Float16)0.0f;
    }
    if (t < H2 * H1) {
        W21h[t]   = (_Float16)W21[t];
        Wobs1h[t] = (_Float16)Wobs1[t];
    }
}

// ---------------------------------------------------------------------------
// Kernel 2: fused MLP + physics. One WG (8 waves) per 32-row batch block.
//   Phase 1: h = relu(x @ W1^T + b1) -> LDS (32x1024 f16)
//   Phase 2: x21/x22 = relu(h @ W^T + b) -> LDS (2 x 32x512 f16)
//            k-outer loop: A loaded once/k-step, B shared by both row-blocks,
//            16 independent WMMA accumulators in flight
//   Phase 3: 512->2 projections (contiguous per-lane chunks, b128 LDS loads,
//            wave-resident weight slices) + physics epilogue
// ---------------------------------------------------------------------------
__global__ __launch_bounds__(256)
void fused_livenet(const float* __restrict__ x,
                   const _Float16* __restrict__ W1h,  const float* __restrict__ b1,
                   const _Float16* __restrict__ W21h, const float* __restrict__ b21,
                   const _Float16* __restrict__ Wobs1h, const float* __restrict__ bobs1,
                   const float* __restrict__ W31,   const float* __restrict__ b31,
                   const float* __restrict__ Wobs2, const float* __restrict__ bobs2,
                   const float* __restrict__ imean, const float* __restrict__ istd,
                   float* __restrict__ out) {
    extern __shared__ _Float16 smem[];
    _Float16* Hs = smem;                    // ROWS x 1024
    _Float16* Xs = smem + ROWS * H1;        // 2 branches x ROWS x 512

    const int lane = threadIdx.x & 31;
    const int wave = threadIdx.x >> 5;
    const int rowBase = blockIdx.x * ROWS;

    const int m  = lane & 15;               // matrix row within a 16-row block
    const int k0 = (lane < 16) ? 0 : 8;     // K-chunk select per WMMA A/B layout

    // ---------------- Phase 1: H block (2 row-blocks) ----------------
    v16h av[2];
    #pragma unroll
    for (int rb = 0; rb < 2; ++rb) {
        const float* xr = x + (size_t)(rowBase + rb * 16 + m) * NUM_IN;
        #pragma unroll
        for (int e = 0; e < 8; ++e) {
            int k = k0 + e;
            av[rb][e] = (k < NUM_IN) ? (_Float16)xr[k] : (_Float16)0.0f;
        }
        #pragma unroll
        for (int e = 0; e < 8; ++e) {
            int k = k0 + 16 + e;
            av[rb][8 + e] = (k < NUM_IN) ? (_Float16)xr[k] : (_Float16)0.0f;
        }
    }
    #pragma unroll 1
    for (int it = 0; it < 8; ++it) {
        int j = wave + 8 * it;              // column tile 0..63
        int n = j * 16 + (lane & 15);
        const v8h* bp = (const v8h*)(W1h + (size_t)n * 32 + k0);
        v8h blo = bp[0];
        v8h bhi = bp[2];                    // +16 halfs
        v16h bv;
        #pragma unroll
        for (int e = 0; e < 8; ++e) { bv[e] = blo[e]; bv[8 + e] = bhi[e]; }
        float bias = b1[n];
        #pragma unroll
        for (int rb = 0; rb < 2; ++rb) {
            v8f c = {};
            c = __builtin_amdgcn_wmma_f32_16x16x32_f16(false, av[rb], false, bv,
                                                       (short)0, c, false, false);
            #pragma unroll
            for (int r = 0; r < 8; ++r) {
                int lrow = rb * 16 + ((lane < 16) ? r : (8 + r));
                float v = c[r] + bias;
                Hs[lrow * H1 + n] = (_Float16)(v > 0.0f ? v : 0.0f);
            }
        }
    }
    __syncthreads();

    // ---------------- Phase 2: x21 / x22 ----------------
    // Each wave owns 8 column tiles (4 of x21, 4 of x22) x 2 row-blocks.
    unsigned off[8];
    float    biasv[8];
    int      ncol[8];
    #pragma unroll
    for (int i = 0; i < 8; ++i) {
        int t = wave + 8 * i;               // 0..63 flat tile; t<32 -> x21 branch
        int n = (t & 31) * 16 + (lane & 15);
        ncol[i]  = n;
        off[i]   = (unsigned)n * H1;        // f16-element offset into weight matrix
        biasv[i] = ((i < 4) ? b21 : bobs1)[n];
    }

    v8f acc[16];
    #pragma unroll
    for (int i = 0; i < 16; ++i) acc[i] = (v8f){};

    const int aoff = m * H1 + k0;
    #pragma unroll 1
    for (int kk = 0; kk < 32; ++kk) {
        const int ab = kk * 32;
        // A fragments for both row-blocks, loaded once per k-step from LDS
        const v8h* ap0 = (const v8h*)(Hs + aoff + ab);
        v8h a0lo = ap0[0], a0hi = ap0[2];
        const v8h* ap1 = (const v8h*)(Hs + 16 * H1 + aoff + ab);
        v8h a1lo = ap1[0], a1hi = ap1[2];
        v16h av0, av1;
        #pragma unroll
        for (int e = 0; e < 8; ++e) {
            av0[e] = a0lo[e]; av0[8 + e] = a0hi[e];
            av1[e] = a1lo[e]; av1[8 + e] = a1hi[e];
        }
        #pragma unroll
        for (int i = 0; i < 8; ++i) {
            const _Float16* wb = (i < 4) ? W21h : Wobs1h;   // uniform base (saddr)
            const v8h* bp = (const v8h*)(wb + off[i] + ab + k0);
            v8h blo = bp[0], bhi = bp[2];
            v16h bv;
            #pragma unroll
            for (int e = 0; e < 8; ++e) { bv[e] = blo[e]; bv[8 + e] = bhi[e]; }
            acc[2 * i]     = __builtin_amdgcn_wmma_f32_16x16x32_f16(
                                 false, av0, false, bv, (short)0, acc[2 * i], false, false);
            acc[2 * i + 1] = __builtin_amdgcn_wmma_f32_16x16x32_f16(
                                 false, av1, false, bv, (short)0, acc[2 * i + 1], false, false);
        }
    }
    // Epilogue: bias + relu -> LDS (f16)
    #pragma unroll
    for (int i = 0; i < 8; ++i) {
        int n = ncol[i];
        _Float16* Xb = Xs + ((i < 4) ? 0 : 1) * (ROWS * H2);
        #pragma unroll
        for (int rb = 0; rb < 2; ++rb) {
            v8f c = acc[2 * i + rb];
            #pragma unroll
            for (int r = 0; r < 8; ++r) {
                int lrow = rb * 16 + ((lane < 16) ? r : (8 + r));
                float v = c[r] + biasv[i];
                Xb[lrow * H2 + n] = (_Float16)(v > 0.0f ? v : 0.0f);
            }
        }
    }
    __syncthreads();

    // ---------------- Phase 3: projections + physics ----------------
    // Lane owns contiguous chunk n in [lane*16, lane*16+16). Weight slices are
    // loaded once per wave and reused for all 4 rows; X rows come in as 2x b128.
    const _Float16* X21 = Xs;
    const _Float16* X22 = Xs + ROWS * H2;
    const int nbase = lane * 16;
    float w31a[16], w31b[16], wo2a[16], wo2b[16];
    #pragma unroll
    for (int e = 0; e < 16; ++e) {
        w31a[e] = W31[nbase + e];
        w31b[e] = W31[H2 + nbase + e];
        wo2a[e] = Wobs2[nbase + e];
        wo2b[e] = Wobs2[H2 + nbase + e];
    }
    #pragma unroll 1
    for (int rr = 0; rr < 4; ++rr) {
        int rl = wave * 4 + rr;             // local row 0..31
        int row = rowBase + rl;
        const v8h* p21 = (const v8h*)(X21 + rl * H2 + nbase);
        v8h c0 = p21[0], c1 = p21[1];
        const v8h* p22 = (const v8h*)(X22 + rl * H2 + nbase);
        v8h d0 = p22[0], d1 = p22[1];
        float a0 = 0.f, a1 = 0.f, a2 = 0.f, a3 = 0.f;
        #pragma unroll
        for (int e = 0; e < 8; ++e) {
            float v21 = (float)c0[e];
            float v22 = (float)d0[e];
            a0 += v21 * w31a[e];
            a1 += v21 * w31b[e];
            a2 += v22 * wo2a[e];
            a3 += v22 * wo2b[e];
        }
        #pragma unroll
        for (int e = 0; e < 8; ++e) {
            float v21 = (float)c1[e];
            float v22 = (float)d1[e];
            a0 += v21 * w31a[8 + e];
            a1 += v21 * w31b[8 + e];
            a2 += v22 * wo2a[8 + e];
            a3 += v22 * wo2b[8 + e];
        }
        #pragma unroll
        for (int off2 = 16; off2 > 0; off2 >>= 1) {
            a0 += __shfl_xor(a0, off2, 32);
            a1 += __shfl_xor(a1, off2, 32);
            a2 += __shfl_xor(a2, off2, 32);
            a3 += __shfl_xor(a3, off2, 32);
        }
        if (lane == 0) {
            float x31_0 = a0 + b31[0];
            float x31_1 = a1 + b31[1];
            float p0 = 4.0f / (1.0f + expf(-(a2 + bobs2[0])));
            float p1 = 4.0f / (1.0f + expf(-(a3 + bobs2[1])));

            const float* xr = x + (size_t)row * NUM_IN;
            float x0v[NUM_IN];
            #pragma unroll
            for (int k = 2; k < NUM_IN; ++k)
                x0v[k] = xr[k] * istd[k] + imean[k];

            float theta = x0v[2], v = x0v[3];
            float st = sinf(theta), ct2 = cosf(theta);
            float ox = x0v[4], oy = x0v[5], oth = x0v[6], ov = x0v[7];
            float dxv = -ox, dyv = -oy;
            float ost = sinf(oth), octv = cosf(oth);
            float barrier = dxv * dxv + dyv * dyv - R2C;
            float bdot = 2.f * dxv * (v * ct2 - ov * octv) + 2.f * dyv * (v * st - ov * ost);
            float Lf2b = 2.f * (v * v + ov * ov - 2.f * v * ov * cosf(theta + oth));
            float Lg1 = -2.f * dxv * v * st + 2.f * dyv * v * ct2;
            float Lg2 = 2.f * dxv * ct2 + 2.f * dyv * st;
            float psum = p0 + p1, pprod = p0 * p1;

            float* o = out + (size_t)row * 14;
            o[0] = x31_0;
            o[1] = x31_1;
            o[2] = -Lg1;
            o[3] = -Lg2;
            o[10] = Lf2b + psum * bdot + pprod * barrier;
            #pragma unroll
            for (int s = 0; s < 3; ++s) {
                float sdx = -x0v[8 + 4 * s];
                float sdy = -x0v[9 + 4 * s];
                float sb = sdx * sdx + sdy * sdy - R2C;
                float sbd = 2.f * sdx * v * ct2 + 2.f * sdy * v * st;
                float g1 = -2.f * sdx * v * st + 2.f * sdy * v * ct2;
                float g2 = 2.f * sdx * ct2 + 2.f * sdy * st;
                o[4 + 2 * s] = -g1;
                o[5 + 2 * s] = -g2;
                o[11 + s] = 2.f * v * v + psum * sbd + pprod * sb;
            }
        }
    }
}

// ---------------------------------------------------------------------------
extern "C" void kernel_launch(void* const* d_in, const int* in_sizes, int n_in,
                              void* d_out, int out_size, void* d_ws, size_t ws_size,
                              hipStream_t stream) {
    const float* x     = (const float*)d_in[0];
    // d_in[1] = sgn (unused by reference forward)
    const float* imean = (const float*)d_in[2];
    const float* istd  = (const float*)d_in[3];
    const float* W1    = (const float*)d_in[4];
    const float* b1    = (const float*)d_in[5];
    const float* W21   = (const float*)d_in[6];
    const float* b21   = (const float*)d_in[7];
    const float* W31   = (const float*)d_in[8];
    const float* b31   = (const float*)d_in[9];
    const float* Wobs1 = (const float*)d_in[10];
    const float* bobs1 = (const float*)d_in[11];
    const float* Wobs2 = (const float*)d_in[12];
    const float* bobs2 = (const float*)d_in[13];
    float* out = (float*)d_out;

    char* ws = (char*)d_ws;
    _Float16* W1h    = (_Float16*)ws;                              // 64 KB
    _Float16* W21h   = (_Float16*)(ws + 65536);                    // 1 MB
    _Float16* Wobs1h = (_Float16*)(ws + 65536 + 1048576);          // 1 MB

    cvt_weights<<<(H2 * H1 + 255) / 256, 256, 0, stream>>>(W1, W21, Wobs1,
                                                           W1h, W21h, Wobs1h);

    size_t smem = (size_t)(ROWS * H1 + 2 * ROWS * H2) * sizeof(_Float16);  // 128 KB
    fused_livenet<<<BB / ROWS, 256, smem, stream>>>(x, W1h, b1, W21h, b21,
                                                    Wobs1h, bobs1, W31, b31,
                                                    Wobs2, bobs2, imean, istd, out);
}
